// CenternetDeconv_50319836840431
// MI455X (gfx1250) — compile-verified, matrix operand order checked by reading, not run on hardware
//
#include <hip/hip_runtime.h>

typedef __attribute__((ext_vector_type(16))) _Float16 v16h;
typedef __attribute__((ext_vector_type(8)))  _Float16 v8h;
typedef __attribute__((ext_vector_type(8)))  float    v8f;

#define BN_EPS 1e-5f

// =====================================================================
// Pack conv weights [Cout][Cin][3][3] (f32) into per-lane WMMA A-fragment
// order (f16): index (((coT*9 + tap)*nCiC + cc)*32 + lane)*16 + j.
// Rows beyond Cout are zero-padded (so 2-wide co tiling is safe).
// =====================================================================
__global__ void pack_w_kernel(const float* __restrict__ W, _Float16* __restrict__ Wp,
                              int Cin, int Cout, int total)
{
    const int t = blockIdx.x * blockDim.x + threadIdx.x;
    if (t >= total) return;
    const int nCiC = Cin >> 5;
    const int lane = t & 31;
    const int rest = t >> 5;
    const int cc   = rest % nCiC;
    const int r2   = rest / nCiC;
    const int tap  = r2 % 9;
    const int coT  = r2 / 9;
    const int half = lane >> 4;
    const int l    = lane & 15;
    const int co   = coT * 16 + l;
    _Float16* out = Wp + (size_t)t * 16;
#pragma unroll
    for (int j = 0; j < 16; ++j) {
        const int kl = (j < 8) ? (half * 8 + j) : (16 + half * 8 + (j - 8));
        const int ci = cc * 32 + kl;
        const float v = (co < Cout) ? W[((size_t)co * Cin + ci) * 9 + tap] : 0.f;
        out[j] = (_Float16)v;
    }
}

// =====================================================================
// Zero the halo row (row index HW) of a pixel-major f16 tensor.
// =====================================================================
__global__ void zrow_kernel(_Float16* __restrict__ row, int count)
{
    const int i = blockIdx.x * blockDim.x + threadIdx.x;
    if (i < count) row[i] = (_Float16)0.f;
}

// =====================================================================
// LDS-tiled transpose+convert: X [C][HW] f32 -> Xt [HW][C] f16.
// =====================================================================
__global__ void nchw_to_pc16_kernel(const float* __restrict__ X,
                                    _Float16* __restrict__ Xt,
                                    int C, int HW)
{
    __shared__ float t[32][33];
    const int p0 = blockIdx.x * 32;
    const int c0 = blockIdx.y * 32;
    const int tx = threadIdx.x, ty = threadIdx.y;   // block (32, 8)
#pragma unroll
    for (int i = 0; i < 4; ++i) {
        const int cl = ty * 4 + i;
        t[cl][tx] = X[(size_t)(c0 + cl) * HW + p0 + tx];
    }
    __syncthreads();
#pragma unroll
    for (int i = 0; i < 4; ++i) {
        const int pl = ty * 4 + i;
        Xt[(size_t)(p0 + pl) * C + c0 + tx] = (_Float16)t[tx][pl];
    }
}

// ---- epilogue helper: store one 8-row x 1-pixel slice of an acc tile ----
__device__ __forceinline__ void store_tile(const v8f& acc, const float* __restrict__ Bs,
                                           float* __restrict__ Y, _Float16* __restrict__ Yt,
                                           int om0, int p, int Cout, int HW, int relu)
{
    if (Y) {
#pragma unroll
        for (int r = 0; r < 8; ++r) {
            const int om = om0 + r;
            if (om < Cout) {
                float v = acc[r] + Bs[om];
                if (relu) v = fmaxf(v, 0.f);
                Y[(size_t)om * HW + p] = v;
            }
        }
    }
    if (Yt) {   // only used when Cout % 32 == 0
        v8h ov;
#pragma unroll
        for (int r = 0; r < 8; ++r) {
            float v = acc[r] + Bs[om0 + r];
            if (relu) v = fmaxf(v, 0.f);
            ov[r] = (_Float16)v;
        }
        *(v8h*)(Yt + (size_t)p * Cout + om0) = ov;   // 16B store
    }
}

#define SHUF16(lo, hi) __builtin_shufflevector(lo, hi, 0,1,2,3,4,5,6,7,8,9,10,11,12,13,14,15)
#define WMMA(a, b, c) __builtin_amdgcn_wmma_f32_16x16x32_f16(false, a, false, b, (short)0, c, false, false)

// =====================================================================
// Implicit-GEMM 3x3 conv (pad 1), 2x2 register tiling:
// one wave32 -> 32(Cout) x 32(pix) tile = 4 accumulators, 4 WMMAs per
// K-chunk on 2 A + 2 B fragments. Out-of-bounds taps read the zero halo
// row at Xt[HW] (branchless, no mask VALU). Block = 4 waves = 128 pixels.
// =====================================================================
__global__ void conv3x3_wmma_kernel(const _Float16* __restrict__ Xt, // [HW+1][Cin]
                                    const _Float16* __restrict__ Wp,
                                    const float* __restrict__ Bs,
                                    float* __restrict__ Y,           // [Cout][HW] or null
                                    _Float16* __restrict__ Yt,       // [HW(+1)][Cout] or null
                                    int Cin, int Cout, int H, int W, int relu)
{
    const int HW   = H * W;
    const int tid  = threadIdx.x & 31;
    const int wave = threadIdx.x >> 5;
    const int half = tid >> 4;
    const int l    = tid & 15;
    const int p0   = (blockIdx.x * 4 + wave) * 32 + l;
    const int p1   = p0 + 16;
    const int nCiC = Cin >> 5;

    const size_t coStride = (size_t)9 * nCiC * 512;     // f16 per coT block
    const _Float16* ap0 = Wp + (size_t)(blockIdx.y * 2) * coStride + (size_t)tid * 16;
    const _Float16* ap1 = ap0 + coStride;

    const int ph0 = p0 / W, pw0 = p0 % W;
    const int ph1 = p1 / W, pw1 = p1 % W;

    v8f acc00 = {}, acc01 = {}, acc10 = {}, acc11 = {};
    for (int tap = 0; tap < 9; ++tap) {
        const int ky = tap / 3, kx = tap - ky * 3;
        const int ih0 = ph0 + ky - 1, iw0 = pw0 + kx - 1;
        const int ih1 = ph1 + ky - 1, iw1 = pw1 + kx - 1;
        const bool in0 = (ih0 >= 0) & (ih0 < H) & (iw0 >= 0) & (iw0 < W);
        const bool in1 = (ih1 >= 0) & (ih1 < H) & (iw1 >= 0) & (iw1 < W);
        const int q0 = in0 ? (ih0 * W + iw0) : HW;      // HW = zero halo row
        const int q1 = in1 ? (ih1 * W + iw1) : HW;
        const _Float16* xr0 = Xt + (size_t)q0 * Cin + half * 8;
        const _Float16* xr1 = Xt + (size_t)q1 * Cin + half * 8;
        for (int cc = 0; cc < nCiC; ++cc) {
            __builtin_prefetch(xr0 + 128, 0, 1);        // global_prefetch_b8
            __builtin_prefetch(xr1 + 128, 0, 1);
            v16h a0 = *(const v16h*)ap0;  ap0 += 512;
            v16h a1 = *(const v16h*)ap1;  ap1 += 512;
            v8h b0lo = *(const v8h*)(xr0);
            v8h b0hi = *(const v8h*)(xr0 + 16);  xr0 += 32;
            v8h b1lo = *(const v8h*)(xr1);
            v8h b1hi = *(const v8h*)(xr1 + 16);  xr1 += 32;
            v16h b0 = SHUF16(b0lo, b0hi);
            v16h b1 = SHUF16(b1lo, b1hi);
            acc00 = WMMA(a0, b0, acc00);
            acc01 = WMMA(a0, b1, acc01);
            acc10 = WMMA(a1, b0, acc10);
            acc11 = WMMA(a1, b1, acc11);
        }
    }
    const int omA = blockIdx.y * 32 + half * 8;
    const int omB = omA + 16;
    store_tile(acc00, Bs, Y, Yt, omA, p0, Cout, HW, relu);
    store_tile(acc01, Bs, Y, Yt, omA, p1, Cout, HW, relu);
    store_tile(acc10, Bs, Y, Yt, omB, p0, Cout, HW, relu);
    store_tile(acc11, Bs, Y, Yt, omB, p1, Cout, HW, relu);
}

// =====================================================================
// WMMA GEMM for the DCN einsum (2x2 tiled): Y[M][Np] = A * Bt^T + bias,
// Bt pixel-major per row (k = tap*Cin + ci), matching packed-A order.
// =====================================================================
__global__ void gemm_wmma_kernel(const _Float16* __restrict__ Bt, // [Np][K]
                                 const _Float16* __restrict__ Wp,
                                 const float* __restrict__ bias,
                                 float* __restrict__ Y,           // [M][Np]
                                 int M, int K, int Np)
{
    const int tid  = threadIdx.x & 31;
    const int wave = threadIdx.x >> 5;
    const int half = tid >> 4;
    const int l    = tid & 15;
    const int p0   = (blockIdx.x * 4 + wave) * 32 + l;
    const int p1   = p0 + 16;
    const int nCh  = K >> 5;

    const size_t coStride = (size_t)nCh * 512;
    const _Float16* ap0 = Wp + (size_t)(blockIdx.y * 2) * coStride + (size_t)tid * 16;
    const _Float16* ap1 = ap0 + coStride;
    const _Float16* br0 = Bt + (size_t)p0 * K + half * 8;
    const _Float16* br1 = Bt + (size_t)p1 * K + half * 8;

    v8f acc00 = {}, acc01 = {}, acc10 = {}, acc11 = {};
    for (int c = 0; c < nCh; ++c) {
        __builtin_prefetch(br0 + 512, 0, 1);
        __builtin_prefetch(br1 + 512, 0, 1);
        v16h a0 = *(const v16h*)ap0;  ap0 += 512;
        v16h a1 = *(const v16h*)ap1;  ap1 += 512;
        v8h b0lo = *(const v8h*)(br0);
        v8h b0hi = *(const v8h*)(br0 + 16);  br0 += 32;
        v8h b1lo = *(const v8h*)(br1);
        v8h b1hi = *(const v8h*)(br1 + 16);  br1 += 32;
        v16h b0 = SHUF16(b0lo, b0hi);
        v16h b1 = SHUF16(b1lo, b1hi);
        acc00 = WMMA(a0, b0, acc00);
        acc01 = WMMA(a0, b1, acc01);
        acc10 = WMMA(a1, b0, acc10);
        acc11 = WMMA(a1, b1, acc11);
    }
    const int omA = blockIdx.y * 32 + half * 8;
    const int omB = omA + 16;
    store_tile(acc00, bias, Y, nullptr, omA, p0, M, Np, 0);
    store_tile(acc01, bias, Y, nullptr, omA, p1, M, Np, 0);
    store_tile(acc10, bias, Y, nullptr, omB, p0, M, Np, 0);
    store_tile(acc11, bias, Y, nullptr, omB, p1, M, Np, 0);
}

// =====================================================================
// Deformable sampling (one image): thread = (tap k, pixel p); writes the
// masked bilinear gather for all ci as f16 into Vt[p][k*C + ci].
// =====================================================================
__global__ void dcn_sample_kernel(const float* __restrict__ Xn,   // [C][H][W] f32
                                  const float* __restrict__ Offn, // [27][H][W] f32
                                  _Float16* __restrict__ Vt,      // [HW][9*C] f16
                                  int C, int H, int W)
{
    const int HW = H * W;
    const int p  = blockIdx.x * blockDim.x + threadIdx.x;
    const int k  = blockIdx.y;   // 0..8
    if (p >= HW) return;
    const int h = p / W, w = p % W;

    const float dy = Offn[(2 * k) * HW + p];
    const float dx = Offn[(2 * k + 1) * HW + p];
    float mk = Offn[(18 + k) * HW + p];
    mk = 1.f / (1.f + __expf(-mk));

    const float py = (float)h - 1.f + (float)(k / 3) + dy;
    const float px = (float)w - 1.f + (float)(k % 3) + dx;
    const float y0f = floorf(py), x0f = floorf(px);
    const float fy = py - y0f, fx = px - x0f;

    int   idx[4];
    float wt[4];
#pragma unroll
    for (int c = 0; c < 4; ++c) {
        const int   cy = c >> 1, cx = c & 1;
        const float yy = y0f + (float)cy, xx = x0f + (float)cx;
        const float wv = (cy ? fy : 1.f - fy) * (cx ? fx : 1.f - fx);
        const bool valid = (yy >= 0.f) && (yy <= (float)(H - 1)) &&
                           (xx >= 0.f) && (xx <= (float)(W - 1));
        const int iy = (int)fminf(fmaxf(yy, 0.f), (float)(H - 1));
        const int ix = (int)fminf(fmaxf(xx, 0.f), (float)(W - 1));
        idx[c] = iy * W + ix;
        wt[c]  = valid ? wv * mk : 0.f;
    }
    _Float16* vrow = Vt + (size_t)p * (9 * C) + (size_t)k * C;
    for (int ci = 0; ci < C; ++ci) {
        const float* xp = Xn + (size_t)ci * HW;
        const float s = xp[idx[0]] * wt[0] + xp[idx[1]] * wt[1] +
                        xp[idx[2]] * wt[2] + xp[idx[3]] * wt[3];
        vrow[ci] = (_Float16)s;
    }
}

// =====================================================================
// Fused BN -> ReLU -> bilinear 2x upsample -> + shortcut (one image).
// =====================================================================
__global__ void bn_up_add_kernel(const float* __restrict__ D,  // [Co][H][W]
                                 const float* __restrict__ s,
                                 const float* __restrict__ bb,
                                 const float* __restrict__ mn,
                                 const float* __restrict__ vr,
                                 const float* __restrict__ SC, // [Co][2H][2W]
                                 float* __restrict__ O,        // [Co][2H][2W]
                                 _Float16* __restrict__ OT,    // [2H*2W(+1)][Co] or null
                                 int Co, int H, int W)
{
    const int H2 = 2 * H, W2 = 2 * W;
    const int p2 = blockIdx.x * blockDim.x + threadIdx.x;
    const int co = blockIdx.y;
    if (p2 >= H2 * W2) return;
    const int h2 = p2 / W2, w2 = p2 % W2;

    const float inv  = s[co] * rsqrtf(vr[co] + BN_EPS);
    const float mu   = mn[co], beta = bb[co];
    const float sy = (float)(H - 1) / (float)(2 * H - 1);
    const float sx = (float)(W - 1) / (float)(2 * W - 1);
    const float cy = (float)h2 * sy, cx = (float)w2 * sx;
    const int iy0 = (int)cy;
    const int ix0 = (int)cx;
    const int iy1 = (iy0 + 1 < H) ? iy0 + 1 : H - 1;
    const int ix1 = (ix0 + 1 < W) ? ix0 + 1 : W - 1;
    const float fy = cy - (float)iy0, fx = cx - (float)ix0;

    const float* dB = D + (size_t)co * H * W;
#define BNR(i) fmaxf((dB[(i)] - mu) * inv + beta, 0.f)
    const float v00 = BNR(iy0 * W + ix0);
    const float v01 = BNR(iy0 * W + ix1);
    const float v10 = BNR(iy1 * W + ix0);
    const float v11 = BNR(iy1 * W + ix1);
#undef BNR
    const float up = v00 * (1.f - fy) * (1.f - fx) + v01 * (1.f - fy) * fx +
                     v10 * fy * (1.f - fx) + v11 * fy * fx;
    const size_t oidx = (size_t)co * H2 * W2 + p2;
    const float outv = up + SC[oidx];
    O[oidx] = outv;
    if (OT) OT[(size_t)p2 * Co + co] = (_Float16)outv;
}

// =====================================================================
// Host orchestration
// =====================================================================
static inline size_t packElems(int Cin, int Cout) {
    return (size_t)((Cout + 31) / 32) * 32 * (size_t)Cin * 9;  // pad to 32 rows
}

extern "C" void kernel_launch(void* const* d_in, const int* in_sizes, int n_in,
                              void* d_out, int out_size, void* d_ws, size_t ws_size,
                              hipStream_t stream) {
    const int N = 4;
    const float* x2 = (const float*)d_in[0];   // (4,256,128,128)
    const float* x3 = (const float*)d_in[1];   // (4,512,64,64)
    const float* x4 = (const float*)d_in[2];   // (4,1024,32,32)
    const float* x5 = (const float*)d_in[3];   // (4,2048,16,16)

    const float* off_w[3]; const float* off_b[3];
    const float* dcn_w[3]; const float* dcn_b[3];
    const float* bn_s[3];  const float* bn_b[3];
    const float* bn_m[3];  const float* bn_v[3];
    for (int i = 0; i < 3; ++i) {
        const int base = 4 + i * 8;
        off_w[i] = (const float*)d_in[base + 0];
        off_b[i] = (const float*)d_in[base + 1];
        dcn_w[i] = (const float*)d_in[base + 2];
        dcn_b[i] = (const float*)d_in[base + 3];
        bn_s[i]  = (const float*)d_in[base + 4];
        bn_b[i]  = (const float*)d_in[base + 5];
        bn_m[i]  = (const float*)d_in[base + 6];
        bn_v[i]  = (const float*)d_in[base + 7];
    }
    const float* sc1_w0 = (const float*)d_in[28]; const float* sc1_b0 = (const float*)d_in[29];
    const float* sc2_w0 = (const float*)d_in[30]; const float* sc2_b0 = (const float*)d_in[31];
    const float* sc2_w1 = (const float*)d_in[32]; const float* sc2_b1 = (const float*)d_in[33];
    const float* sc3_w0 = (const float*)d_in[34]; const float* sc3_b0 = (const float*)d_in[35];
    const float* sc3_w1 = (const float*)d_in[36]; const float* sc3_b1 = (const float*)d_in[37];
    const float* sc3_w2 = (const float*)d_in[38]; const float* sc3_b2 = (const float*)d_in[39];
    (void)in_sizes; (void)n_in; (void)out_size; (void)ws_size;

    // ---- workspace carve ----
    float* ws = (float*)d_ws;
    _Float16* wpool = (_Float16*)ws;
    size_t f16o = 0;
    auto alloc16 = [&](int cin, int cout) {
        _Float16* r = wpool + f16o; f16o += packElems(cin, cout); return r;
    };
    _Float16* wp_off1 = alloc16(2048, 27);
    _Float16* wp_dcn1 = alloc16(2048, 256);
    _Float16* wp_off2 = alloc16(256, 27);
    _Float16* wp_dcn2 = alloc16(256, 128);
    _Float16* wp_off3 = alloc16(128, 27);
    _Float16* wp_dcn3 = alloc16(128, 64);
    _Float16* wp_sc1  = alloc16(1024, 256);
    _Float16* wp_s2a  = alloc16(512, 128);
    _Float16* wp_s2b  = alloc16(128, 128);
    _Float16* wp_s3a  = alloc16(256, 64);
    _Float16* wp_s3b  = alloc16(64, 64);
    _Float16* wp_s3c  = alloc16(64, 64);

    size_t o = (f16o + 1) / 2;                              // floats used so far
    _Float16* xt   = (_Float16*)(ws + o); o += 2099200;     // max transp. + halo slack
    _Float16* ytA  = (_Float16*)(ws + o); o += 526336;
    _Float16* ytB  = (_Float16*)(ws + o); o += 526336;
    float*    scF  = ws + o; o += 1048576;
    float*    offF = ws + o; o += 110592;
    _Float16* Vt   = (_Float16*)(ws + o); o += 2359296;
    float*    dcnF = ws + o; o += 262144;
    float*    xs1F = ws + o; o += 262144;
    _Float16* xs1T = (_Float16*)(ws + o); o += 133120;
    float*    xs2F = ws + o; o += 524288;
    _Float16* xs2T = (_Float16*)(ws + o); o += 264192;
    (void)o;

    // ---- pack all weights once ----
    auto packLaunch = [&](const float* w, _Float16* dst, int cin, int cout) {
        const int total = (int)(packElems(cin, cout) / 16);
        pack_w_kernel<<<dim3((total + 255) / 256), dim3(256), 0, stream>>>(w, dst, cin, cout, total);
    };
    packLaunch(off_w[0], wp_off1, 2048, 27);
    packLaunch(dcn_w[0], wp_dcn1, 2048, 256);
    packLaunch(off_w[1], wp_off2, 256, 27);
    packLaunch(dcn_w[1], wp_dcn2, 256, 128);
    packLaunch(off_w[2], wp_off3, 128, 27);
    packLaunch(dcn_w[2], wp_dcn3, 128, 64);
    packLaunch(sc1_w0, wp_sc1, 1024, 256);
    packLaunch(sc2_w0, wp_s2a, 512, 128);
    packLaunch(sc2_w1, wp_s2b, 128, 128);
    packLaunch(sc3_w0, wp_s3a, 256, 64);
    packLaunch(sc3_w1, wp_s3b, 64, 64);
    packLaunch(sc3_w2, wp_s3c, 64, 64);

    auto zrow = [&](_Float16* buf, int HW, int C) {
        zrow_kernel<<<dim3((C + 255) / 256), dim3(256), 0, stream>>>(buf + (size_t)HW * C, C);
    };

    const dim3 cblk(128);
    const dim3 tblk(32, 8);

    for (int n = 0; n < N; ++n) {
        const float* x5n = x5 + (size_t)n * 2048 * 256;
        const float* x4n = x4 + (size_t)n * 1024 * 1024;
        const float* x3n = x3 + (size_t)n * 512 * 4096;
        const float* x2n = x2 + (size_t)n * 256 * 16384;
        float* outn = (float*)d_out + (size_t)n * 64 * 16384;

        // ---------------- Layer 1 ----------------
        nchw_to_pc16_kernel<<<dim3(256/32, 2048/32), tblk, 0, stream>>>(x5n, xt, 2048, 256);
        zrow(xt, 256, 2048);
        conv3x3_wmma_kernel<<<dim3(256/128, 1), cblk, 0, stream>>>(
            xt, wp_off1, off_b[0], offF, nullptr, 2048, 27, 16, 16, 0);
        dcn_sample_kernel<<<dim3(1, 9), dim3(256), 0, stream>>>(x5n, offF, Vt, 2048, 16, 16);
        gemm_wmma_kernel<<<dim3(256/128, 8), cblk, 0, stream>>>(
            Vt, wp_dcn1, dcn_b[0], dcnF, 256, 2048 * 9, 256);
        nchw_to_pc16_kernel<<<dim3(1024/32, 1024/32), tblk, 0, stream>>>(x4n, xt, 1024, 1024);
        zrow(xt, 1024, 1024);
        conv3x3_wmma_kernel<<<dim3(1024/128, 8), cblk, 0, stream>>>(
            xt, wp_sc1, sc1_b0, scF, nullptr, 1024, 256, 32, 32, 0);
        bn_up_add_kernel<<<dim3(1024/256, 256), dim3(256), 0, stream>>>(
            dcnF, bn_s[0], bn_b[0], bn_m[0], bn_v[0], scF, xs1F, xs1T, 256, 16, 16);
        zrow(xs1T, 1024, 256);

        // ---------------- Layer 2 ----------------
        conv3x3_wmma_kernel<<<dim3(1024/128, 1), cblk, 0, stream>>>(
            xs1T, wp_off2, off_b[1], offF, nullptr, 256, 27, 32, 32, 0);
        dcn_sample_kernel<<<dim3(1024/256, 9), dim3(256), 0, stream>>>(xs1F, offF, Vt, 256, 32, 32);
        gemm_wmma_kernel<<<dim3(1024/128, 4), cblk, 0, stream>>>(
            Vt, wp_dcn2, dcn_b[1], dcnF, 128, 256 * 9, 1024);
        nchw_to_pc16_kernel<<<dim3(4096/32, 512/32), tblk, 0, stream>>>(x3n, xt, 512, 4096);
        zrow(xt, 4096, 512);
        conv3x3_wmma_kernel<<<dim3(4096/128, 4), cblk, 0, stream>>>(
            xt, wp_s2a, sc2_b0, nullptr, ytA, 512, 128, 64, 64, 1);
        zrow(ytA, 4096, 128);
        conv3x3_wmma_kernel<<<dim3(4096/128, 4), cblk, 0, stream>>>(
            ytA, wp_s2b, sc2_b1, scF, nullptr, 128, 128, 64, 64, 0);
        bn_up_add_kernel<<<dim3(4096/256, 128), dim3(256), 0, stream>>>(
            dcnF, bn_s[1], bn_b[1], bn_m[1], bn_v[1], scF, xs2F, xs2T, 128, 32, 32);
        zrow(xs2T, 4096, 128);

        // ---------------- Layer 3 ----------------
        conv3x3_wmma_kernel<<<dim3(4096/128, 1), cblk, 0, stream>>>(
            xs2T, wp_off3, off_b[2], offF, nullptr, 128, 27, 64, 64, 0);
        dcn_sample_kernel<<<dim3(4096/256, 9), dim3(256), 0, stream>>>(xs2F, offF, Vt, 128, 64, 64);
        gemm_wmma_kernel<<<dim3(4096/128, 2), cblk, 0, stream>>>(
            Vt, wp_dcn3, dcn_b[2], dcnF, 64, 128 * 9, 4096);
        nchw_to_pc16_kernel<<<dim3(16384/32, 256/32), tblk, 0, stream>>>(x2n, xt, 256, 16384);
        zrow(xt, 16384, 256);
        conv3x3_wmma_kernel<<<dim3(16384/128, 2), cblk, 0, stream>>>(
            xt, wp_s3a, sc3_b0, nullptr, ytA, 256, 64, 128, 128, 1);
        zrow(ytA, 16384, 64);
        conv3x3_wmma_kernel<<<dim3(16384/128, 2), cblk, 0, stream>>>(
            ytA, wp_s3b, sc3_b1, nullptr, ytB, 64, 64, 128, 128, 1);
        zrow(ytB, 16384, 64);
        conv3x3_wmma_kernel<<<dim3(16384/128, 2), cblk, 0, stream>>>(
            ytB, wp_s3c, sc3_b2, scF, nullptr, 64, 64, 128, 128, 0);
        bn_up_add_kernel<<<dim3(16384/256, 64), dim3(256), 0, stream>>>(
            dcnF, bn_s[2], bn_b[2], bn_m[2], bn_v[2], scF, outn, nullptr, 64, 64, 64);
    }
}
